// character_aware_nlm_42846593744990
// MI455X (gfx1250) — matrix operand order repaired and checked by reading, not compiled, not run on hardware
//
#include <hip/hip_runtime.h>
#include <math.h>

// ---------------- model dims ----------------
#define BB   32
#define TT   128
#define LL   21
#define EMB  15
#define MT   (BB*TT)        // 4096 rows
#define TOT  1100
#define TOTP 1120           // pad to mult of 32
#define HID  650
#define HIDP 672            // pad to mult of 32
#define G4   2600
#define G4P  2624           // pad to mult of 32
#define VOC  32000

typedef __bf16 bf16_t;
typedef __attribute__((ext_vector_type(16))) __bf16 bf16x16;
typedef __attribute__((ext_vector_type(8)))  __bf16 bf16x8;
typedef __attribute__((ext_vector_type(8)))  float  v8f;

__device__ __forceinline__ bf16x16 cat8(bf16x8 lo, bf16x8 hi) {
  return __builtin_shufflevector(lo, hi, 0,1,2,3,4,5,6,7,8,9,10,11,12,13,14,15);
}
__device__ __forceinline__ float sigf(float x) { return 1.0f / (1.0f + expf(-x)); }

// ---------------- elementwise helpers ----------------
__global__ void zero_f32_kernel(float* p, long long n) {
  for (long long i = blockIdx.x * (long long)blockDim.x + threadIdx.x; i < n;
       i += (long long)gridDim.x * blockDim.x) p[i] = 0.0f;
}
__global__ void zero_bf16_kernel(bf16_t* p, long long n) {
  for (long long i = blockIdx.x * (long long)blockDim.x + threadIdx.x; i < n;
       i += (long long)gridDim.x * blockDim.x) p[i] = (bf16_t)0.0f;
}
// fp32 [srows x scols] -> bf16 [drows x dcols] zero-padded
__global__ void pad_cvt_kernel(bf16_t* __restrict__ dst, const float* __restrict__ src,
                               int srows, int scols, int drows, int dcols) {
  long long n = (long long)drows * dcols;
  for (long long i = blockIdx.x * (long long)blockDim.x + threadIdx.x; i < n;
       i += (long long)gridDim.x * blockDim.x) {
    int r = (int)(i / dcols), c = (int)(i % dcols);
    float v = (r < srows && c < scols) ? src[(long long)r * scols + c] : 0.0f;
    dst[i] = (bf16_t)v;
  }
}
// dst[i] = a[i] (+ b[i]) for i<n else 0, length npad
__global__ void pad_bias_kernel(float* __restrict__ dst, const float* __restrict__ a,
                                const float* __restrict__ b, int n, int npad) {
  for (int i = blockIdx.x * blockDim.x + threadIdx.x; i < npad; i += gridDim.x * blockDim.x)
    dst[i] = (i < n) ? (a[i] + (b ? b[i] : 0.0f)) : 0.0f;
}

// ---------------- char CNN (embed gather + conv + tanh + maxpool) ----------------
__global__ __launch_bounds__(128)
void embed_cnn_kernel(const int* __restrict__ ids, const float* __restrict__ emb,
                      const float* __restrict__ w1, const float* __restrict__ b1,
                      const float* __restrict__ w2, const float* __restrict__ b2,
                      const float* __restrict__ w3, const float* __restrict__ b3,
                      const float* __restrict__ w4, const float* __restrict__ b4,
                      const float* __restrict__ w5, const float* __restrict__ b5,
                      const float* __restrict__ w6, const float* __restrict__ b6,
                      const float* __restrict__ w7, const float* __restrict__ b7,
                      float* __restrict__ z) {
  __shared__ float e[LL * EMB];
  const int bt = blockIdx.x;
  const int* row = ids + bt * LL;
  for (int i = threadIdx.x; i < LL * EMB; i += blockDim.x) {
    int cpos = i / EMB, cdim = i % EMB;
    e[i] = emb[(long long)row[cpos] * EMB + cdim];
  }
  __syncthreads();
  for (int f = threadIdx.x; f < TOTP; f += blockDim.x) {
    if (f >= TOT) { z[(long long)bt * TOTP + f] = 0.0f; continue; }
    int w, o; const float* W; const float* Bv;
    if      (f <  50) { w = 1; o = f;        W = w1; Bv = b1; }
    else if (f < 150) { w = 2; o = f - 50;   W = w2; Bv = b2; }
    else if (f < 300) { w = 3; o = f - 150;  W = w3; Bv = b3; }
    else if (f < 500) { w = 4; o = f - 300;  W = w4; Bv = b4; }
    else if (f < 700) { w = 5; o = f - 500;  W = w5; Bv = b5; }
    else if (f < 900) { w = 6; o = f - 700;  W = w6; Bv = b6; }
    else              { w = 7; o = f - 900;  W = w7; Bv = b7; }
    const float* Wr = W + (long long)o * w * EMB;
    float bias = Bv[o];
    float m = -1e30f;
    for (int p = 0; p <= LL - w; ++p) {
      float s = 0.0f;
      for (int i = 0; i < w; ++i)
        for (int d = 0; d < EMB; ++d)
          s += e[(p + i) * EMB + d] * Wr[i * EMB + d];
      float v = tanhf(s + bias);
      m = fmaxf(m, v);
    }
    z[(long long)bt * TOTP + f] = m;
  }
}

// ---------------- bf16 WMMA GEMM: C[MxN] = A[MxK] * Bw[NxK]^T + bias ----------------
// Wave tile: (MTILES*16) x 32.  M multiple of MTILES*16; N,K multiples of 32.
// MTILES=4 -> 8 WMMAs per k-step over 12 b128 loads: 4x B-fragment reuse.
template<int MTILES>
__global__ __launch_bounds__(256)
void gemm_bf16_wmma(float* __restrict__ C, const bf16_t* __restrict__ A,
                    const bf16_t* __restrict__ Bw, const float* __restrict__ bias,
                    int M, int N, int K, int ldc) {
  const int lane = threadIdx.x & 31;
  const int wave = threadIdx.x >> 5;
  const int ntn  = N >> 5;
  const int ntiles = (M / (16 * MTILES)) * ntn;
  int tile = blockIdx.x * 8 + wave;
  if (tile >= ntiles) return;            // wave-uniform: EXEC stays all-ones
  const int tm = tile / ntn;
  const int tn = tile % ntn;
  const int half = lane >> 4;
  const int l15  = lane & 15;

  const bf16_t* aRow[MTILES];
#pragma unroll
  for (int i = 0; i < MTILES; ++i)
    aRow[i] = A + (size_t)(tm * 16 * MTILES + i * 16 + l15) * K;
  const bf16_t* bRow0 = Bw + (size_t)(tn * 32 + l15) * K;   // tile cols 0..15
  const bf16_t* bRow1 = bRow0 + (size_t)16 * K;             // tile cols 16..31

  v8f acc0[MTILES], acc1[MTILES];
#pragma unroll
  for (int i = 0; i < MTILES; ++i) { acc0[i] = (v8f){}; acc1[i] = (v8f){}; }

  for (int k0 = 0; k0 < K; k0 += 32) {
    const int ka = k0 + half * 16;    // A frag: K = ka..ka+15 (contiguous)
    const int kb = k0 + half * 16;    // B frag: K = kb..kb+15 (contiguous)
    bf16x16 b0 = cat8(*(const bf16x8*)(bRow0 + kb), *(const bf16x8*)(bRow0 + kb + 8));
    bf16x16 b1 = cat8(*(const bf16x8*)(bRow1 + kb), *(const bf16x8*)(bRow1 + kb + 8));
#pragma unroll
    for (int i = 0; i < MTILES; ++i) {
      bf16x16 a = cat8(*(const bf16x8*)(aRow[i] + ka), *(const bf16x8*)(aRow[i] + ka + 8));
      acc0[i] = __builtin_amdgcn_wmma_f32_16x16x32_bf16(false, a, false, b0, (short)0, acc0[i], false, false);
      acc1[i] = __builtin_amdgcn_wmma_f32_16x16x32_bf16(false, a, false, b1, (short)0, acc1[i], false, false);
    }
  }
  const int col0 = tn * 32 + l15;                 // C: row = half*8+r, col = l15
  float bs0 = bias ? bias[col0]      : 0.0f;
  float bs1 = bias ? bias[col0 + 16] : 0.0f;
#pragma unroll
  for (int i = 0; i < MTILES; ++i) {
#pragma unroll
    for (int r = 0; r < 8; ++r) {
      int row = tm * 16 * MTILES + i * 16 + half * 8 + r;
      C[(size_t)row * ldc + col0]      = acc0[i][r] + bs0;
      C[(size_t)row * ldc + col0 + 16] = acc1[i][r] + bs1;
    }
  }
}

// ---------------- highway combine: z = t*relu(H) + (1-t)*z ----------------
__global__ void highway_combine_kernel(float* __restrict__ z, const float* __restrict__ H,
                                       const float* __restrict__ Tg) {
  long long n = (long long)MT * TOT;
  for (long long i = blockIdx.x * (long long)blockDim.x + threadIdx.x; i < n;
       i += (long long)gridDim.x * blockDim.x) {
    int r = (int)(i / TOT), c = (int)(i % TOT);
    long long off = (long long)r * TOTP + c;
    float t = sigf(Tg[off]);
    float h = fmaxf(H[off], 0.0f);
    z[off] = t * h + (1.0f - t) * z[off];
  }
}

// ---------------- LSTM cell (one time step, all batches) ----------------
__global__ __launch_bounds__(256)
void lstm_cell_kernel(const float* __restrict__ xih,   // [MT x G4P], row = b*TT + t
                      const float* __restrict__ gates, // [BB x G4P]  (h @ Whh^T)
                      float* __restrict__ c,           // [BB x HIDP]
                      bf16_t* __restrict__ hbf,        // [BB x HIDP]
                      bf16_t* __restrict__ hs,         // [MT x HIDP] or null
                      int t) {
  const int b = blockIdx.x;
  const long long xrow = (long long)(b * TT + t) * G4P;
  const long long grow = (long long)b * G4P;
  for (int j = threadIdx.x; j < HID; j += blockDim.x) {
    float gi = xih[xrow + j]            + gates[grow + j];
    float gf = xih[xrow + HID + j]      + gates[grow + HID + j];
    float gg = xih[xrow + 2 * HID + j]  + gates[grow + 2 * HID + j];
    float go = xih[xrow + 3 * HID + j]  + gates[grow + 3 * HID + j];
    float cc = sigf(gf) * c[(long long)b * HIDP + j] + sigf(gi) * tanhf(gg);
    float hh = sigf(go) * tanhf(cc);
    c[(long long)b * HIDP + j] = cc;
    hbf[(long long)b * HIDP + j] = (bf16_t)hh;
    if (hs) hs[(long long)(b * TT + t) * HIDP + j] = (bf16_t)hh;
  }
}

// ---------------- host side ----------------
static inline char* bump(char*& p, size_t bytes) {
  char* r = p;
  p += (bytes + 255) & ~(size_t)255;
  return r;
}
static inline int nblk(long long n, int thr) {
  long long b = (n + thr - 1) / thr;
  if (b > 16384) b = 16384;
  if (b < 1) b = 1;
  return (int)b;
}
static void launch_gemm(hipStream_t s, float* C, const bf16_t* A, const bf16_t* B,
                        const float* bias, int M, int N, int K, int ldc) {
  if ((M & 63) == 0) {
    int ntiles = (M / 64) * (N / 32);
    gemm_bf16_wmma<4><<<(ntiles + 7) / 8, 256, 0, s>>>(C, A, B, bias, M, N, K, ldc);
  } else {
    int ntiles = (M / 32) * (N / 32);
    gemm_bf16_wmma<2><<<(ntiles + 7) / 8, 256, 0, s>>>(C, A, B, bias, M, N, K, ldc);
  }
}

extern "C" void kernel_launch(void* const* d_in, const int* in_sizes, int n_in,
                              void* d_out, int out_size, void* d_ws, size_t ws_size,
                              hipStream_t stream) {
  const int*   char_ids = (const int*)d_in[0];
  const float* char_emb = (const float*)d_in[1];
  const float *cw[7], *cb[7];
  for (int i = 0; i < 7; ++i) { cw[i] = (const float*)d_in[2 + 2 * i]; cb[i] = (const float*)d_in[3 + 2 * i]; }
  const float* hwWh[2] = { (const float*)d_in[16], (const float*)d_in[20] };
  const float* hwbh[2] = { (const float*)d_in[17], (const float*)d_in[21] };
  const float* hwWt[2] = { (const float*)d_in[18], (const float*)d_in[22] };
  const float* hwbt[2] = { (const float*)d_in[19], (const float*)d_in[23] };
  const float* Wih[2]  = { (const float*)d_in[24], (const float*)d_in[28] };
  const float* Whh[2]  = { (const float*)d_in[25], (const float*)d_in[29] };
  const float* bih[2]  = { (const float*)d_in[26], (const float*)d_in[30] };
  const float* bhh[2]  = { (const float*)d_in[27], (const float*)d_in[31] };
  const float* outW    = (const float*)d_in[32];
  const float* outb    = (const float*)d_in[33];
  float* out = (float*)d_out;

  // ---- workspace layout (bump allocated) ----
  char* p = (char*)d_ws;
  bf16_t* wWhB[2]; bf16_t* wWtB[2];
  for (int l = 0; l < 2; ++l) {
    wWhB[l] = (bf16_t*)bump(p, (size_t)TOTP * TOTP * 2);
    wWtB[l] = (bf16_t*)bump(p, (size_t)TOTP * TOTP * 2);
  }
  bf16_t* wIhB[2]; bf16_t* wHhB[2];
  wIhB[0] = (bf16_t*)bump(p, (size_t)G4P * TOTP * 2);
  wIhB[1] = (bf16_t*)bump(p, (size_t)G4P * HIDP * 2);
  wHhB[0] = (bf16_t*)bump(p, (size_t)G4P * HIDP * 2);
  wHhB[1] = (bf16_t*)bump(p, (size_t)G4P * HIDP * 2);
  bf16_t* wOutB = (bf16_t*)bump(p, (size_t)VOC * HIDP * 2);
  float* bhP[2]; float* btP[2]; float* blP[2];
  for (int l = 0; l < 2; ++l) {
    bhP[l] = (float*)bump(p, (size_t)TOTP * 4);
    btP[l] = (float*)bump(p, (size_t)TOTP * 4);
    blP[l] = (float*)bump(p, (size_t)G4P * 4);
  }
  float*  z    = (float*) bump(p, (size_t)MT * TOTP * 4);
  bf16_t* zb   = (bf16_t*)bump(p, (size_t)MT * TOTP * 2);
  float*  Hbuf = (float*) bump(p, (size_t)MT * TOTP * 4);
  float*  Tbuf = (float*) bump(p, (size_t)MT * TOTP * 4);
  float*  xih  = (float*) bump(p, (size_t)MT * G4P * 4);
  float*  gat  = (float*) bump(p, (size_t)BB * G4P * 4);
  float*  cst  = (float*) bump(p, (size_t)BB * HIDP * 4);
  bf16_t* hbf  = (bf16_t*)bump(p, (size_t)BB * HIDP * 2);
  bf16_t* hs0  = (bf16_t*)bump(p, (size_t)MT * HIDP * 2);

  // ---- weight / bias conversion (bf16, zero-padded) ----
  for (int l = 0; l < 2; ++l) {
    pad_cvt_kernel<<<nblk((long long)TOTP * TOTP, 256), 256, 0, stream>>>(wWhB[l], hwWh[l], TOT, TOT, TOTP, TOTP);
    pad_cvt_kernel<<<nblk((long long)TOTP * TOTP, 256), 256, 0, stream>>>(wWtB[l], hwWt[l], TOT, TOT, TOTP, TOTP);
    pad_bias_kernel<<<nblk(TOTP, 256), 256, 0, stream>>>(bhP[l], hwbh[l], nullptr, TOT, TOTP);
    pad_bias_kernel<<<nblk(TOTP, 256), 256, 0, stream>>>(btP[l], hwbt[l], nullptr, TOT, TOTP);
    pad_bias_kernel<<<nblk(G4P, 256), 256, 0, stream>>>(blP[l], bih[l], bhh[l], G4, G4P);
  }
  pad_cvt_kernel<<<nblk((long long)G4P * TOTP, 256), 256, 0, stream>>>(wIhB[0], Wih[0], G4, TOT, G4P, TOTP);
  pad_cvt_kernel<<<nblk((long long)G4P * HIDP, 256), 256, 0, stream>>>(wIhB[1], Wih[1], G4, HID, G4P, HIDP);
  pad_cvt_kernel<<<nblk((long long)G4P * HIDP, 256), 256, 0, stream>>>(wHhB[0], Whh[0], G4, HID, G4P, HIDP);
  pad_cvt_kernel<<<nblk((long long)G4P * HIDP, 256), 256, 0, stream>>>(wHhB[1], Whh[1], G4, HID, G4P, HIDP);
  pad_cvt_kernel<<<nblk((long long)VOC * HIDP, 256), 256, 0, stream>>>(wOutB, outW, VOC, HID, VOC, HIDP);

  // ---- char CNN ----
  embed_cnn_kernel<<<MT, 128, 0, stream>>>(char_ids, char_emb,
      cw[0], cb[0], cw[1], cb[1], cw[2], cb[2], cw[3], cb[3],
      cw[4], cb[4], cw[5], cb[5], cw[6], cb[6], z);

  // ---- highway layers ----
  for (int l = 0; l < 2; ++l) {
    pad_cvt_kernel<<<nblk((long long)MT * TOTP, 256), 256, 0, stream>>>(zb, z, MT, TOTP, MT, TOTP);
    launch_gemm(stream, Hbuf, zb, wWhB[l], bhP[l], MT, TOTP, TOTP, TOTP);
    launch_gemm(stream, Tbuf, zb, wWtB[l], btP[l], MT, TOTP, TOTP, TOTP);
    highway_combine_kernel<<<nblk((long long)MT * TOT, 256), 256, 0, stream>>>(z, Hbuf, Tbuf);
  }

  // ---- LSTM layer 0 ----
  pad_cvt_kernel<<<nblk((long long)MT * TOTP, 256), 256, 0, stream>>>(zb, z, MT, TOTP, MT, TOTP);
  launch_gemm(stream, xih, zb, wIhB[0], blP[0], MT, G4P, TOTP, G4P);
  zero_bf16_kernel<<<nblk((long long)BB * HIDP, 256), 256, 0, stream>>>(hbf, (long long)BB * HIDP);
  zero_f32_kernel<<<nblk((long long)BB * HIDP, 256), 256, 0, stream>>>(cst, (long long)BB * HIDP);
  zero_bf16_kernel<<<nblk((long long)MT * HIDP, 256), 256, 0, stream>>>(hs0, (long long)MT * HIDP);
  for (int t = 0; t < TT; ++t) {
    launch_gemm(stream, gat, hbf, wHhB[0], nullptr, BB, G4P, HIDP, G4P);
    lstm_cell_kernel<<<BB, 256, 0, stream>>>(xih, gat, cst, hbf, hs0, t);
  }

  // ---- LSTM layer 1 ----
  launch_gemm(stream, xih, hs0, wIhB[1], blP[1], MT, G4P, HIDP, G4P);
  zero_bf16_kernel<<<nblk((long long)BB * HIDP, 256), 256, 0, stream>>>(hbf, (long long)BB * HIDP);
  zero_f32_kernel<<<nblk((long long)BB * HIDP, 256), 256, 0, stream>>>(cst, (long long)BB * HIDP);
  for (int t = 0; t < TT; ++t) {
    launch_gemm(stream, gat, hbf, wHhB[1], nullptr, BB, G4P, HIDP, G4P);
    lstm_cell_kernel<<<BB, 256, 0, stream>>>(xih, gat, cst, hbf, nullptr, t);
  }

  // ---- output projection: last h (t=127) @ out_W^T + out_b -> d_out (32 x 32000) ----
  launch_gemm(stream, out, hbf, wOutB, outb, BB, VOC, HIDP, VOC);
}
